// AttentiveFP_super_9835475108416
// MI455X (gfx1250) — matrix operand order repaired and checked by reading, not compiled, not run on hardware
//
#include <hip/hip_runtime.h>
#include <hip/hip_bf16.h>
#include <math.h>

typedef _Float16 h16;
typedef __attribute__((ext_vector_type(16))) _Float16 v16h;
typedef __attribute__((ext_vector_type(8)))  float    v8f;

// ---------------------------------------------------------------- utilities

__device__ __forceinline__ float leaky_f(float v) { return v > 0.0f ? v : 0.01f * v; }

// order-preserving float -> uint key (for atomicMax-based segment max)
__device__ __forceinline__ unsigned fkey(float f) {
    unsigned b = __float_as_uint(f);
    return (b & 0x80000000u) ? ~b : (b | 0x80000000u);
}
__device__ __forceinline__ float funkey(unsigned u) {
    unsigned b = (u & 0x80000000u) ? (u ^ 0x80000000u) : ~u;
    return __uint_as_float(b);
}

// CDNA5 async global -> LDS copy (16B per lane), tracked by ASYNCcnt
__device__ __forceinline__ void async_copy_b128(unsigned lds_off, const void* gaddr) {
    asm volatile("global_load_async_to_lds_b128 %0, %1, off"
                 :: "v"(lds_off), "v"(gaddr) : "memory");
}
__device__ __forceinline__ void wait_async_le1() {
    asm volatile("s_wait_asynccnt 0x1" ::: "memory");
}
__device__ __forceinline__ void wait_async_le0() {
    asm volatile("s_wait_asynccnt 0x0" ::: "memory");
}

// ---------------------------------------------------------------- small kernels

__global__ void fill_f32_kernel(float* p, float v, int n) {
    int i = blockIdx.x * blockDim.x + threadIdx.x;
    if (i < n) p[i] = v;
}
__global__ void fill_u32_kernel(unsigned* p, unsigned v, int n) {
    int i = blockIdx.x * blockDim.x + threadIdx.x;
    if (i < n) p[i] = v;
}

// convert f32 matrix [Mr x K] (row stride rowStride, starting col colOff) into
// f16 [? x Kp] zero-padded rows/cols (Kp multiple of 32) -> unguarded WMMA loads
__global__ void convh_kernel(const float* __restrict__ W, h16* __restrict__ out,
                             int Mr, int K, int Kp, int rowStride, int colOff, int total) {
    int i = blockIdx.x * blockDim.x + threadIdx.x;
    if (i >= total) return;
    int n = i / Kp, k = i - n * Kp;
    float v = (n < Mr && k < K) ? W[(size_t)n * rowStride + colOff + k] : 0.0f;
    out[i] = (h16)v;
}

// out[n] = dot(V[n, 0:64], a)
__global__ void rowdot_kernel(const float* __restrict__ V, const float* __restrict__ a,
                              float* __restrict__ out, int rows) {
    int n = blockIdx.x * blockDim.x + threadIdx.x;
    if (n >= rows) return;
    const float4* vp = (const float4*)(V + (size_t)n * 64);
    const float4* ap = (const float4*)a;
    float s = 0.0f;
#pragma unroll
    for (int i = 0; i < 16; ++i) {
        float4 v = vp[i], w = ap[i];
        s += v.x * w.x + v.y * w.y + v.z * w.z + v.w * w.w;
    }
    out[n] = s;
}

__global__ void relu_kernel(float* p, int n) {
    int i = blockIdx.x * blockDim.x + threadIdx.x;
    if (i < n) p[i] = fmaxf(p[i], 0.0f);
}

// p[n,f] = elu(p[n,f] + bias[f])   (64 features)
__global__ void elu_bias_kernel(float* p, const float* __restrict__ bias, int n) {
    int i = blockIdx.x * blockDim.x + threadIdx.x;
    if (i >= n) return;
    float v = p[i] + bias[i & 63];
    p[i] = v > 0.0f ? v : (expf(v) - 1.0f);
}

// GRU combine: out = relu((1-z)*n + z*h) over [rows x 64]; gi/gh are [rows x 192]
__global__ void gru_combine_kernel(const float* __restrict__ gi, const float* __restrict__ gh,
                                   const float* __restrict__ hprev, float* __restrict__ out,
                                   int count) {
    int i = blockIdx.x * blockDim.x + threadIdx.x;
    if (i >= count) return;
    int n = i >> 6, f = i & 63;
    size_t b = (size_t)n * 192 + f;
    float ir = gi[b], iz = gi[b + 64], inn = gi[b + 128];
    float hr = gh[b], hz = gh[b + 64], hn = gh[b + 128];
    float r = 1.0f / (1.0f + expf(-(ir + hr)));
    float z = 1.0f / (1.0f + expf(-(iz + hz)));
    float nn = tanhf(inn + r * hn);
    float o = (1.0f - z) * nn + z * hprev[i];
    out[i] = fmaxf(o, 0.0f);
}

// logit[e] = leaky(asrc[s] + adst[d]);  atomicMax segment key.  src==null -> s=e
__global__ void logit_kernel(const float* __restrict__ asrc, const float* __restrict__ adst,
                             const int* __restrict__ src, const int* __restrict__ dst,
                             float* __restrict__ logit, unsigned* __restrict__ segmax, int E) {
    int e = blockIdx.x * blockDim.x + threadIdx.x;
    if (e >= E) return;
    int s = src ? src[e] : e;
    int d = dst[e];
    float lg = leaky_f(asrc[s] + adst[d]);
    logit[e] = lg;
    atomicMax(&segmax[d], fkey(lg));
}

// expv[e] = exp(logit[e] - max[dst]);  atomicAdd into segsum
__global__ void softmax_exp_kernel(const float* __restrict__ logit, const int* __restrict__ dst,
                                   const unsigned* __restrict__ segmax, float* __restrict__ expv,
                                   float* __restrict__ segsum, int E) {
    int e = blockIdx.x * blockDim.x + threadIdx.x;
    if (e >= E) return;
    int d = dst[e];
    float ev = expf(logit[e] - funkey(segmax[d]));
    expv[e] = ev;
    atomicAdd(&segsum[d], ev);
}

// wave-per-edge: out[dst] += msg[src] * (expv[e]/(segsum[dst]+eps)); 2 feats/lane
__global__ __launch_bounds__(256)
void agg_kernel(const float* __restrict__ msg, const int* __restrict__ src,
                const int* __restrict__ dst, const float* __restrict__ expv,
                const float* __restrict__ segsum, float* __restrict__ out, int E) {
    int wave = threadIdx.x >> 5, lane = threadIdx.x & 31;
    int e = blockIdx.x * 8 + wave;
    if (e >= E) return;
    int s = src ? src[e] : e;
    int d = dst[e];
    float a = expv[e] / (segsum[d] + 1e-16f);
    const float2* mp = (const float2*)(msg + (size_t)s * 64);
    float2 v = mp[lane];
    atomicAdd(&out[(size_t)d * 64 + lane * 2 + 0], v.x * a);
    atomicAdd(&out[(size_t)d * 64 + lane * 2 + 1], v.y * a);
}

// wave-per-row: out[idx[n]] += v[n]  (64-wide rows)
__global__ __launch_bounds__(256)
void scatter_add_kernel(const float* __restrict__ v, const int* __restrict__ idx,
                        float* __restrict__ out, int rows) {
    int wave = threadIdx.x >> 5, lane = threadIdx.x & 31;
    int n = blockIdx.x * 8 + wave;
    if (n >= rows) return;
    int d = idx[n];
    const float2* mp = (const float2*)(v + (size_t)n * 64);
    float2 val = mp[lane];
    atomicAdd(&out[(size_t)d * 64 + lane * 2 + 0], val.x);
    atomicAdd(&out[(size_t)d * 64 + lane * 2 + 1], val.y);
}

// gate-conv edge logit: hj = leaky(xa[src] + W1b @ ea); p = dot(att_l, hj);
// logit = leaky(p + xr[dst]); wave per edge
__global__ __launch_bounds__(256)
void gate_alpha_kernel(const float* __restrict__ xa, const float* __restrict__ xr,
                       const int* __restrict__ src, const int* __restrict__ dst,
                       const float* __restrict__ edge_attr, const float* __restrict__ g_lin1_w,
                       const float* __restrict__ att_l, float* __restrict__ logit,
                       unsigned* __restrict__ segmax, int E) {
    __shared__ float Wb[64 * 10];
    __shared__ float Al[64];
    for (int i = threadIdx.x; i < 640; i += 256) {
        int f = i / 10, j = i - f * 10;
        Wb[i] = g_lin1_w[f * 74 + 64 + j];
    }
    for (int i = threadIdx.x; i < 64; i += 256) Al[i] = att_l[i];
    __syncthreads();

    int wave = threadIdx.x >> 5, lane = threadIdx.x & 31;
    int e = blockIdx.x * 8 + wave;
    if (e >= E) return;
    int s = src[e], d = dst[e];
    float ea[10];
#pragma unroll
    for (int j = 0; j < 10; ++j) ea[j] = edge_attr[(size_t)e * 10 + j];
    float p = 0.0f;
#pragma unroll
    for (int t = 0; t < 2; ++t) {
        int f = lane * 2 + t;
        float hj = xa[(size_t)s * 64 + f];
#pragma unroll
        for (int j = 0; j < 10; ++j) hj += Wb[f * 10 + j] * ea[j];
        hj = leaky_f(hj);
        p += hj * Al[f];
    }
#pragma unroll
    for (int o = 16; o > 0; o >>= 1) p += __shfl_xor(p, o, 32);
    if (lane == 0) {
        float lg = leaky_f(p + xr[d]);
        logit[e] = lg;
        atomicMax(&segmax[d], fkey(lg));
    }
}

// cbuf[g, 0:192] = [d3[g,0:128] | outg[g,0:64]]
__global__ void concat_kernel(const float* __restrict__ d3, const float* __restrict__ outg,
                              float* __restrict__ cbuf, int count) {
    int i = blockIdx.x * blockDim.x + threadIdx.x;
    if (i >= count) return;
    int g = i / 192, j = i - g * 192;
    cbuf[i] = (j < 128) ? d3[(size_t)g * 128 + j] : outg[(size_t)g * 64 + (j - 128)];
}

// final: out[g] = dot(c2o[g], w) + b
__global__ void final_kernel(const float* __restrict__ c2o, const float* __restrict__ w,
                             const float* __restrict__ b, float* __restrict__ out, int G) {
    int g = blockIdx.x * blockDim.x + threadIdx.x;
    if (g >= G) return;
    const float4* vp = (const float4*)(c2o + (size_t)g * 64);
    const float4* wp = (const float4*)w;
    float s = 0.0f;
#pragma unroll
    for (int i = 0; i < 16; ++i) {
        float4 v = vp[i], ww = wp[i];
        s += v.x * ww.x + v.y * ww.y + v.z * ww.z + v.w * ww.w;
    }
    out[g] = s + b[0];
}

// ---------------------------------------------------------------- WMMA GEMM
// C[M,N] = act((A @ B^T + bias) * scale + shift)
// A: padded f16 [Mp x Kp]  (Mp = ceil64(M), Kp = ceil32(K), zero-filled)
// B: padded f16 [N  x Kp]  (weight rows w[n][k])
// Block: 256 threads = 8 waves; 64x64 C tile; wave -> 1 m-tile x 2 n-tiles.
// A tile staged via double-buffered GLOBAL_LOAD_ASYNC_TO_LDS_B128 (ASYNCcnt);
// last iteration peeled so the steady-state loop body is branch-free.
__global__ __launch_bounds__(256)
void gemm_kernel(const h16* __restrict__ A, const h16* __restrict__ B,
                 const float* __restrict__ bias, const float* __restrict__ scale,
                 const float* __restrict__ shift, float* __restrict__ C,
                 int M, int N, int Kp, int act) {
    __shared__ h16 As[2][64 * 40];  // 2 x (64 rows x 32 k), stride 40 halfs
    const int tid  = threadIdx.x;
    const int wave = tid >> 5;
    const int lane = tid & 31;
    const int l16  = lane & 15;
    const int hi   = lane >> 4;
    const int mt   = wave & 3;
    const int ntB  = (wave >> 2) * 2;  // 0 or 2
    const int m0   = blockIdx.x * 64;
    const int n0   = blockIdx.y * 64;

    // staging assignment: 4 threads per row, 8 halfs (16B) each
    const int srow = tid >> 2;
    const int sseg = (tid & 3) * 8;
    const h16* agp = A + (size_t)(m0 + srow) * Kp + sseg;  // + 32*ki per step
    const unsigned lds0 = (unsigned)(size_t)&As[0][srow * 40 + sseg];
    const unsigned lds1 = (unsigned)(size_t)&As[1][srow * 40 + sseg];

    v8f acc0 = {0, 0, 0, 0, 0, 0, 0, 0};
    v8f acc1 = {0, 0, 0, 0, 0, 0, 0, 0};

    // one k-step: A frag from LDS buffer (ki&1), B frags from global, 2 WMMAs
    auto compute = [&](int ki) {
        const h16* cbase = &As[ki & 1][(mt * 16 + l16) * 40];
        union { v16h v; unsigned u[8]; } af;
#pragma unroll
        for (int q = 0; q < 8; ++q) {
            int k = hi * 16 + 2 * q;
            af.u[q] = *(const unsigned*)(cbase + k);
        }
        union { v16h v; unsigned u[8]; } bf0, bf1;
        {
            int n = n0 + ntB * 16 + l16;
            int nn = (n < N) ? n : 0;
            const h16* brow = B + (size_t)nn * Kp + ki * 32 + hi * 16;
#pragma unroll
            for (int q = 0; q < 8; ++q) bf0.u[q] = *(const unsigned*)(brow + 2 * q);
        }
        {
            int n = n0 + (ntB + 1) * 16 + l16;
            int nn = (n < N) ? n : 0;
            const h16* brow = B + (size_t)nn * Kp + ki * 32 + hi * 16;
#pragma unroll
            for (int q = 0; q < 8; ++q) bf1.u[q] = *(const unsigned*)(brow + 2 * q);
        }
        acc0 = __builtin_amdgcn_wmma_f32_16x16x32_f16(false, af.v, false, bf0.v,
                                                      (short)0, acc0, false, false);
        acc1 = __builtin_amdgcn_wmma_f32_16x16x32_f16(false, af.v, false, bf1.v,
                                                      (short)0, acc1, false, false);
    };

    const int nk = Kp >> 5;
    // prologue: prefetch tile 0 into buffer 0
    async_copy_b128(lds0, agp);

    // steady state: branch-free body, prefetch tile ki+1 while computing ki
    for (int ki = 0; ki < nk - 1; ++ki) {
        async_copy_b128((ki & 1) ? lds0 : lds1, agp + (size_t)(ki + 1) * 32);
        wait_async_le1();  // async loads complete in order -> tile ki resident
        __syncthreads();
        compute(ki);
        __syncthreads();   // all reads of buffer (ki&1) done before overwrite
    }
    // peeled last tile
    wait_async_le0();
    __syncthreads();
    compute(nk - 1);

    // epilogue: bias + BN scale/shift + activation (guards only here)
#pragma unroll
    for (int t = 0; t < 2; ++t) {
        int col = n0 + (ntB + t) * 16 + l16;
        if (col >= N) continue;
        float bv = bias  ? bias[col]  : 0.0f;
        float sc = scale ? scale[col] : 1.0f;
        float sh = shift ? shift[col] : 0.0f;
#pragma unroll
        for (int r = 0; r < 8; ++r) {
            int row = m0 + mt * 16 + hi * 8 + r;
            if (row >= M) continue;
            float v = (t == 0 ? acc0[r] : acc1[r]) + bv;
            v = v * sc + sh;
            if (act == 1) v = fmaxf(v, 0.0f);
            else if (act == 2) v = leaky_f(v);
            C[(size_t)row * N + col] = v;
        }
    }
}

// ---------------------------------------------------------------- host

extern "C" void kernel_launch(void* const* d_in, const int* in_sizes, int n_in,
                              void* d_out, int out_size, void* d_ws, size_t ws_size,
                              hipStream_t stream) {
    (void)n_in; (void)out_size; (void)ws_size;
    // inputs (setup_inputs dict order, params flattened in insertion order)
    const float* x_in      = (const float*)d_in[0];
    const int*   edge_idx  = (const int*)d_in[1];
    const float* edge_attr = (const float*)d_in[2];
    const float* descr     = (const float*)d_in[3];
    const int*   batch     = (const int*)d_in[4];
    const float* p_lin1_w  = (const float*)d_in[5];
    const float* p_lin1_b  = (const float*)d_in[6];
    const float* p_glin1_w = (const float*)d_in[7];
    const float* p_gattl   = (const float*)d_in[8];
    const float* p_gattr   = (const float*)d_in[9];
    const float* p_glin2_w = (const float*)d_in[10];
    const float* p_gbias   = (const float*)d_in[11];
    const float* p_g0_wih  = (const float*)d_in[12];
    const float* p_g0_whh  = (const float*)d_in[13];
    const float* p_g0_bih  = (const float*)d_in[14];
    const float* p_g0_bhh  = (const float*)d_in[15];
    const float* p_gat_w   = (const float*)d_in[16];
    const float* p_gat_as  = (const float*)d_in[17];
    const float* p_gat_ad  = (const float*)d_in[18];
    const float* p_gat_b   = (const float*)d_in[19];
    const float* p_gru_wih = (const float*)d_in[20];
    const float* p_gru_whh = (const float*)d_in[21];
    const float* p_gru_bih = (const float*)d_in[22];
    const float* p_gru_bhh = (const float*)d_in[23];
    const float* p_mol_w   = (const float*)d_in[24];
    const float* p_mol_as  = (const float*)d_in[25];
    const float* p_mol_ad  = (const float*)d_in[26];
    const float* p_mol_b   = (const float*)d_in[27];
    const float* p_mg_wih  = (const float*)d_in[28];
    const float* p_mg_whh  = (const float*)d_in[29];
    const float* p_mg_bih  = (const float*)d_in[30];
    const float* p_mg_bhh  = (const float*)d_in[31];
    const float* p_m1_w = (const float*)d_in[32]; const float* p_m1_b = (const float*)d_in[33];
    const float* p_bn1g = (const float*)d_in[34]; const float* p_bn1b = (const float*)d_in[35];
    const float* p_m2_w = (const float*)d_in[36]; const float* p_m2_b = (const float*)d_in[37];
    const float* p_bn2g = (const float*)d_in[38]; const float* p_bn2b = (const float*)d_in[39];
    const float* p_m3_w = (const float*)d_in[40]; const float* p_m3_b = (const float*)d_in[41];
    const float* p_bn3g = (const float*)d_in[42]; const float* p_bn3b = (const float*)d_in[43];
    const float* p_c1_w = (const float*)d_in[44]; const float* p_c1_b = (const float*)d_in[45];
    const float* p_c2_w = (const float*)d_in[46]; const float* p_c2_b = (const float*)d_in[47];
    const float* p_c3_w = (const float*)d_in[48]; const float* p_c3_b = (const float*)d_in[49];

    const int N  = in_sizes[4];
    const int E  = in_sizes[1] / 2;
    const int D  = 200;
    const int G  = in_sizes[3] / D;
    const int L  = in_sizes[16] / (64 * 64);
    const int IN = in_sizes[5] / 64;
    const int* e_src = edge_idx;
    const int* e_dst = edge_idx + E;

    // workspace carve-out
    char* wsp = (char*)d_ws;
    size_t cur = 0;
    auto alloc = [&](size_t bytes) -> char* {
        cur = (cur + 255) & ~(size_t)255;
        char* p = wsp + cur;
        cur += bytes;
        return p;
    };
    float*    x0     = (float*)alloc((size_t)N * 64 * 4);
    float*    hb     = (float*)alloc((size_t)N * 64 * 4);
    float*    ns1    = (float*)alloc((size_t)N * 64 * 4);
    float*    ns2    = (float*)alloc((size_t)N * 64 * 4);
    float*    giA    = (float*)alloc((size_t)N * 192 * 4);
    float*    ghB    = (float*)alloc((size_t)N * 192 * 4);
    float*    asrcB  = (float*)alloc((size_t)N * 4);
    float*    adstB  = (float*)alloc((size_t)N * 4);
    float*    logit  = (float*)alloc((size_t)E * 4);
    float*    expv   = (float*)alloc((size_t)E * 4);
    unsigned* segmax = (unsigned*)alloc((size_t)N * 4);
    float*    segsum = (float*)alloc((size_t)N * 4);
    float*    outg   = (float*)alloc((size_t)G * 64 * 4);
    float*    hmol   = (float*)alloc((size_t)G * 64 * 4);
    float*    adstg  = (float*)alloc((size_t)G * 4);
    float*    d1     = (float*)alloc((size_t)G * 400 * 4);
    float*    d2     = (float*)alloc((size_t)G * 256 * 4);
    float*    d3     = (float*)alloc((size_t)G * 128 * 4);
    float*    cbuf   = (float*)alloc((size_t)G * 192 * 4);
    float*    c1o    = (float*)alloc((size_t)G * 128 * 4);
    float*    c2o    = (float*)alloc((size_t)G * 64 * 4);

    // shared padded-f16 A-operand staging buffer (max over all gemms)
    size_t afN = (size_t)((N + 63) & ~63) * 64;       // node-level A (K<=64)
    size_t afG = (size_t)((G + 63) & ~63) * 448;      // graph-level A (K<=416)
    h16* Af = (h16*)alloc((afN > afG ? afN : afG) * 2);

    auto cdiv = [](int a, int b) { return (a + b - 1) / b; };
    auto fillf = [&](float* p, float v, int n) {
        fill_f32_kernel<<<cdiv(n, 256), 256, 0, stream>>>(p, v, n);
    };
    auto fillu = [&](unsigned* p, unsigned v, int n) {
        fill_u32_kernel<<<cdiv(n, 256), 256, 0, stream>>>(p, v, n);
    };
    // f32 weight [Nr x K] (stride, colOff) -> padded f16 [Nr x ceil32(K)]
    auto convw = [&](const float* W, int Nr, int K, int rowStride, int colOff) -> h16* {
        int Kp = (K + 31) & ~31;
        h16* o = (h16*)alloc((size_t)Nr * Kp * 2);
        int tot = Nr * Kp;
        convh_kernel<<<cdiv(tot, 256), 256, 0, stream>>>(W, o, Nr, K, Kp, rowStride, colOff, tot);
        return o;
    };
    auto gemm = [&](const float* A, const h16* B, const float* bias, const float* scale,
                    const float* shift, float* C, int M, int Nc, int K, int act) {
        int Kp = (K + 31) & ~31;
        int Mp = (M + 63) & ~63;
        int tot = Mp * Kp;
        convh_kernel<<<cdiv(tot, 256), 256, 0, stream>>>(A, Af, M, K, Kp, K, 0, tot);
        dim3 g(Mp / 64, cdiv(Nc, 64));
        gemm_kernel<<<g, 256, 0, stream>>>(Af, B, bias, scale, shift, C, M, Nc, Kp, act);
    };

    // ---- weight conversion (f32 -> padded f16) --------------------------
    h16* W_lin1 = convw(p_lin1_w, 64, IN, IN, 0);
    h16* W_g1a  = convw(p_glin1_w, 64, 64, 74, 0);   // first 64 cols of g_lin1_w
    h16* W_gl2  = convw(p_glin2_w, 64, 64, 64, 0);
    h16* W_g0ih = convw(p_g0_wih, 192, 64, 64, 0);
    h16* W_g0hh = convw(p_g0_whh, 192, 64, 64, 0);
    h16* W_gat[8]; h16* W_ih[8]; h16* W_hh[8];
    for (int l = 0; l < L; ++l) {
        W_gat[l] = convw(p_gat_w + (size_t)l * 64 * 64, 64, 64, 64, 0);
        W_ih[l]  = convw(p_gru_wih + (size_t)l * 192 * 64, 192, 64, 64, 0);
        W_hh[l]  = convw(p_gru_whh + (size_t)l * 192 * 64, 192, 64, 64, 0);
    }
    h16* W_mol = convw(p_mol_w, 64, 64, 64, 0);
    h16* W_mih = convw(p_mg_wih, 192, 64, 64, 0);
    h16* W_mhh = convw(p_mg_whh, 192, 64, 64, 0);
    h16* W_m1  = convw(p_m1_w, 2 * D, D, D, 0);
    h16* W_m2  = convw(p_m2_w, 256, 2 * D, 2 * D, 0);
    h16* W_m3  = convw(p_m3_w, 128, 256, 256, 0);
    h16* W_c1  = convw(p_c1_w, 128, 192, 192, 0);
    h16* W_c2  = convw(p_c2_w, 64, 128, 128, 0);

    // ---- x = leaky(x @ lin1_w.T + lin1_b) -------------------------------
    gemm(x_in, W_lin1, p_lin1_b, nullptr, nullptr, x0, N, 64, IN, 2);

    // ---- gate_conv + GRU0 -----------------------------------------------
    gemm(x0, W_g1a, nullptr, nullptr, nullptr, ns1, N, 64, 64, 0);  // xa
    gemm(x0, W_gl2, nullptr, nullptr, nullptr, ns2, N, 64, 64, 0);  // mm
    rowdot_kernel<<<cdiv(N, 256), 256, 0, stream>>>(x0, p_gattr, asrcB, N);  // xr
    fillu(segmax, 0u, N);
    fillf(segsum, 0.0f, N);
    gate_alpha_kernel<<<cdiv(E, 8), 256, 0, stream>>>(ns1, asrcB, e_src, e_dst, edge_attr,
                                                      p_glin1_w, p_gattl, logit, segmax, E);
    softmax_exp_kernel<<<cdiv(E, 256), 256, 0, stream>>>(logit, e_dst, segmax, expv, segsum, E);
    fillf(hb, 0.0f, N * 64);
    agg_kernel<<<cdiv(E, 8), 256, 0, stream>>>(ns2, e_src, e_dst, expv, segsum, hb, E);
    elu_bias_kernel<<<cdiv(N * 64, 256), 256, 0, stream>>>(hb, p_gbias, N * 64);
    gemm(hb, W_g0ih, p_g0_bih, nullptr, nullptr, giA, N, 192, 64, 0);
    gemm(x0, W_g0hh, p_g0_bhh, nullptr, nullptr, ghB, N, 192, 64, 0);
    gru_combine_kernel<<<cdiv(N * 64, 256), 256, 0, stream>>>(giA, ghB, x0, x0, N * 64);

    // ---- GAT layers + GRU ------------------------------------------------
    for (int l = 0; l < L; ++l) {
        gemm(x0, W_gat[l], nullptr, nullptr, nullptr, ns1, N, 64, 64, 0);  // hs
        rowdot_kernel<<<cdiv(N, 256), 256, 0, stream>>>(ns1, p_gat_as + (size_t)l * 64, asrcB, N);
        rowdot_kernel<<<cdiv(N, 256), 256, 0, stream>>>(ns1, p_gat_ad + (size_t)l * 64, adstB, N);
        fillu(segmax, 0u, N);
        fillf(segsum, 0.0f, N);
        logit_kernel<<<cdiv(E, 256), 256, 0, stream>>>(asrcB, adstB, e_src, e_dst, logit, segmax, E);
        softmax_exp_kernel<<<cdiv(E, 256), 256, 0, stream>>>(logit, e_dst, segmax, expv, segsum, E);
        fillf(hb, 0.0f, N * 64);
        agg_kernel<<<cdiv(E, 8), 256, 0, stream>>>(ns1, e_src, e_dst, expv, segsum, hb, E);
        elu_bias_kernel<<<cdiv(N * 64, 256), 256, 0, stream>>>(hb, p_gat_b + (size_t)l * 64, N * 64);
        gemm(hb, W_ih[l], p_gru_bih + (size_t)l * 192, nullptr, nullptr, giA, N, 192, 64, 0);
        gemm(x0, W_hh[l], p_gru_bhh + (size_t)l * 192, nullptr, nullptr, ghB, N, 192, 64, 0);
        gru_combine_kernel<<<cdiv(N * 64, 256), 256, 0, stream>>>(giA, ghB, x0, x0, N * 64);
    }

    // ---- readout: out = relu(segment_sum(x, batch)) ----------------------
    fillf(outg, 0.0f, G * 64);
    scatter_add_kernel<<<cdiv(N, 8), 256, 0, stream>>>(x0, batch, outg, N);
    relu_kernel<<<cdiv(G * 64, 256), 256, 0, stream>>>(outg, G * 64);

    // hoisted (x0 fixed during readout): hs = x0 @ mol_w.T, asrc
    gemm(x0, W_mol, nullptr, nullptr, nullptr, ns1, N, 64, 64, 0);
    rowdot_kernel<<<cdiv(N, 256), 256, 0, stream>>>(ns1, p_mol_as, asrcB, N);

    for (int t = 0; t < 2; ++t) {
        gemm(outg, W_mol, nullptr, nullptr, nullptr, hmol, G, 64, 64, 0);  // hd (temp in hmol)
        rowdot_kernel<<<cdiv(G, 256), 256, 0, stream>>>(hmol, p_mol_ad, adstg, G);
        fillu(segmax, 0u, G);
        fillf(segsum, 0.0f, G);
        logit_kernel<<<cdiv(N, 256), 256, 0, stream>>>(asrcB, adstg, nullptr, batch, logit, segmax, N);
        softmax_exp_kernel<<<cdiv(N, 256), 256, 0, stream>>>(logit, batch, segmax, expv, segsum, N);
        fillf(hmol, 0.0f, G * 64);
        agg_kernel<<<cdiv(N, 8), 256, 0, stream>>>(ns1, nullptr, batch, expv, segsum, hmol, N);
        elu_bias_kernel<<<cdiv(G * 64, 256), 256, 0, stream>>>(hmol, p_mol_b, G * 64);
        gemm(hmol, W_mih, p_mg_bih, nullptr, nullptr, giA, G, 192, 64, 0);
        gemm(outg, W_mhh, p_mg_bhh, nullptr, nullptr, ghB, G, 192, 64, 0);
        gru_combine_kernel<<<cdiv(G * 64, 256), 256, 0, stream>>>(giA, ghB, outg, outg, G * 64);
    }

    // ---- meta_fc ---------------------------------------------------------
    gemm(descr, W_m1, p_m1_b, p_bn1g, p_bn1b, d1, G, 2 * D, D, 1);
    gemm(d1,    W_m2, p_m2_b, p_bn2g, p_bn2b, d2, G, 256, 2 * D, 1);
    gemm(d2,    W_m3, p_m3_b, p_bn3g, p_bn3b, d3, G, 128, 256, 1);

    // ---- head ------------------------------------------------------------
    concat_kernel<<<cdiv(G * 192, 256), 256, 0, stream>>>(d3, outg, cbuf, G * 192);
    gemm(cbuf, W_c1, p_c1_b, nullptr, nullptr, c1o, G, 128, 192, 1);
    gemm(c1o,  W_c2, p_c2_b, nullptr, nullptr, c2o, G, 64, 128, 1);
    final_kernel<<<cdiv(G, 256), 256, 0, stream>>>(c2o, p_c3_w, p_c3_b, (float*)d_out, G);
}
